// ShallowGCNet_10926396801126
// MI455X (gfx1250) — matrix operand order repaired and checked by reading, not compiled, not run on hardware
//
#include <hip/hip_runtime.h>
#include <hip/hip_bf16.h>
#include <stdint.h>

// ---------------------------------------------------------------------------
// Types for CDNA5 WMMA (wave32): 16x16x32 bf16 -> f32 accumulate
// ---------------------------------------------------------------------------
typedef __attribute__((ext_vector_type(16))) __bf16 bf16x16;
typedef __attribute__((ext_vector_type(8)))  float  f32x8;

// Native f32 -> bf16 (compiler emits V_CVT_PK_BF16_F32 pairs)
__device__ __forceinline__ unsigned short f2bf(float f) {
    __bf16 h = (__bf16)f;
    return __builtin_bit_cast(unsigned short, h);
}

__device__ __forceinline__ unsigned hash_u32(unsigned x) {
    x ^= x >> 16; x *= 0x7feb352dU;
    x ^= x >> 15; x *= 0x846ca68bU;
    x ^= x >> 16; return x;
}

// ---------------------------------------------------------------------------
// WMMA GEMM: Out[n x outW] = f(A[n x Fin]) @ W[Fin x Fout] (+ bias)
//   Wt is bf16 bits, transposed: Wt[col][k], col-padded to a multiple of 16.
//   MODE 0: identity A     (layer 1)
//   MODE 1: leaky-relu A   (layers 2-4: activation of previous conv fused here)
//   MODE 2: leaky-relu + dropout(0.5)*2 on A, bias added at store (final layer)
//   Each wave -> 16 rows x NT*16 cols. 8 waves/block -> 128 rows/block.
// ---------------------------------------------------------------------------
template <int NT, int MODE>
__global__ __launch_bounds__(256)
void gcn_gemm_wmma(const float* __restrict__ A,
                   const unsigned short* __restrict__ Wt,
                   float* __restrict__ Out,
                   const float* __restrict__ bias,
                   int n, int Fin, int outW, int outStride)
{
    const int lane = threadIdx.x & 31;
    const int wave = threadIdx.x >> 5;
    const int rt   = blockIdx.x * (blockDim.x >> 5) + wave;
    if (rt * 16 >= n) return;                 // uniform per wave: EXEC stays all-1
    const int row0 = rt * 16;
    const int col0 = blockIdx.y * (NT * 16);
    const int m    = lane & 15;
    const int half = lane >> 4;

    f32x8 acc[NT];
#pragma unroll
    for (int t = 0; t < NT; ++t)
#pragma unroll
        for (int j = 0; j < 8; ++j) acc[t][j] = 0.0f;

    int arow = row0 + m; if (arow >= n) arow = n - 1;
    const float* ap = A + (size_t)arow * Fin + half * 8;

    for (int k0 = 0; k0 < Fin; k0 += 32) {
        // A fragment: lane needs K = [k0+half*8, +8) in VGPR0-3, +16 in VGPR4-7
        float4 a0 = *(const float4*)(ap + k0);
        float4 a1 = *(const float4*)(ap + k0 + 4);
        float4 a2 = *(const float4*)(ap + k0 + 16);
        float4 a3 = *(const float4*)(ap + k0 + 20);
        float av[16] = {a0.x,a0.y,a0.z,a0.w, a1.x,a1.y,a1.z,a1.w,
                        a2.x,a2.y,a2.z,a2.w, a3.x,a3.y,a3.z,a3.w};
        bf16x16 afv;
#pragma unroll
        for (int i = 0; i < 16; ++i) {
            float v = av[i];
            if (MODE >= 1) v = (v > 0.0f) ? v : 0.01f * v;     // leaky relu
            if (MODE == 2) {                                    // dropout p=0.5, x2
                unsigned kk = (unsigned)(k0 + half * 8 + (i < 8 ? i : i + 8));
                unsigned h  = hash_u32(((unsigned)arow * (unsigned)Fin + kk) ^ 42u);
                v = (h & 1u) ? v * 2.0f : 0.0f;
            }
            afv[i] = (__bf16)v;                                 // native cvt
        }
#pragma unroll
        for (int t = 0; t < NT; ++t) {
            // B fragment: lane = col (l&15), K = [k0 + half*8, +8) then +16, matching A
            const unsigned short* bp =
                Wt + (size_t)(col0 + t * 16 + m) * Fin + half * 8 + k0;
            union { uint4 q[2]; bf16x16 v; } bf_;
            bf_.q[0] = *(const uint4*)(bp);
            bf_.q[1] = *(const uint4*)(bp + 16);
            acc[t] = __builtin_amdgcn_wmma_f32_16x16x32_bf16(
                false, afv, false, bf_.v, (short)0, acc[t], false, false);
        }
    }

    // C/D layout: col = lane&15, row = half*8 + r (VGPR r)
#pragma unroll
    for (int t = 0; t < NT; ++t) {
        const int c = col0 + t * 16 + m;
        if (c >= outW) continue;
        const float badd = bias ? bias[c] : 0.0f;
#pragma unroll
        for (int r = 0; r < 8; ++r) {
            const int rr = row0 + half * 8 + r;
            if (rr < n) Out[(size_t)rr * outStride + c] = acc[t][r] + badd;
        }
    }
}

// ---------------------------------------------------------------------------
// Weight prep: Wt[col][k] (bf16 bits) from W[k][col] (f32), cols padded w/ zeros
// ---------------------------------------------------------------------------
__global__ void k_wconvert(const float* __restrict__ W, unsigned short* __restrict__ Wt,
                           int Fin, int Fout, int FoutPad)
{
    int i = blockIdx.x * blockDim.x + threadIdx.x;
    int total = FoutPad * Fin;
    if (i >= total) return;
    int c = i / Fin, k = i % Fin;
    float v = (c < Fout) ? W[(size_t)k * Fout + c] : 0.0f;
    Wt[i] = f2bf(v);
}

// ---------------------------------------------------------------------------
// Graph-norm precompute (done once, reused by all 4 conv layers)
// ---------------------------------------------------------------------------
__global__ void k_deg_init(float* deg, int n) {
    int i = blockIdx.x * blockDim.x + threadIdx.x;
    if (i < n) deg[i] = 1.0f;                       // self loop
}
__global__ void k_deg_count(const long long* __restrict__ dst, float* deg, int e) {
    int i = blockIdx.x * blockDim.x + threadIdx.x;
    if (i < e) atomicAdd(&deg[(int)dst[i]], 1.0f);
}
__global__ void k_dinv(float* deg, int n) {
    int i = blockIdx.x * blockDim.x + threadIdx.x;
    if (i < n) deg[i] = rsqrtf(deg[i]);             // deg>0 always (self loop)
}
__global__ void k_sc(const float* __restrict__ dinv, float* sc, int n) {
    int i = blockIdx.x * blockDim.x + threadIdx.x;
    if (i < n) sc[i] = dinv[i] * dinv[i];           // self-loop coefficient
}
__global__ void k_wnorm(const float* __restrict__ dinv,
                        const long long* __restrict__ src,
                        const long long* __restrict__ dst,
                        float* wn, int e) {
    int i = blockIdx.x * blockDim.x + threadIdx.x;
    if (i < e) wn[i] = dinv[(int)src[i]] * dinv[(int)dst[i]];
}

// ---------------------------------------------------------------------------
// Aggregation: out[i] = bias + sc[i]*h[i] (init), then += wn[e]*h[src] per edge
// ---------------------------------------------------------------------------
__global__ void k_agg_init(const float* __restrict__ h, const float* __restrict__ sc,
                           const float* __restrict__ bias, float* __restrict__ out,
                           int n, int F)
{
    int i = blockIdx.x * blockDim.x + threadIdx.x;
    int fq = F >> 2;
    if (i >= n * fq) return;
    int row = i / fq, c = (i % fq) * 4;
    float s = sc[row];
    float4 hv = *(const float4*)(h + (size_t)row * F + c);
    float4 o;
    o.x = bias[c + 0] + s * hv.x;
    o.y = bias[c + 1] + s * hv.y;
    o.z = bias[c + 2] + s * hv.z;
    o.w = bias[c + 3] + s * hv.w;
    *(float4*)(out + (size_t)row * F + c) = o;
}

__global__ void k_agg_edges(const float* __restrict__ h, const float* __restrict__ wn,
                            const long long* __restrict__ src,
                            const long long* __restrict__ dst,
                            float* __restrict__ out, int e, int F)
{
    long long i = (long long)blockIdx.x * blockDim.x + threadIdx.x;
    int fq = F >> 2;
    if (i >= (long long)e * fq) return;
    int ed = (int)(i / fq), c = (int)(i % fq) * 4;
    int s = (int)src[ed], d = (int)dst[ed];
    float w = wn[ed];
    float4 hv = *(const float4*)(h + (size_t)s * F + c);
    float* o = out + (size_t)d * F + c;
    atomicAdd(o + 0, w * hv.x);
    atomicAdd(o + 1, w * hv.y);
    atomicAdd(o + 2, w * hv.z);
    atomicAdd(o + 3, w * hv.w);
}

// ---------------------------------------------------------------------------
// Host driver
// ---------------------------------------------------------------------------
extern "C" void kernel_launch(void* const* d_in, const int* in_sizes, int n_in,
                              void* d_out, int out_size, void* d_ws, size_t ws_size,
                              hipStream_t stream)
{
    (void)n_in; (void)out_size; (void)ws_size;
    const float*     x   = (const float*)d_in[0];
    const long long* ei  = (const long long*)d_in[1];   // int64 edge_index [2,E]
    const float* W1 = (const float*)d_in[2];  const float* b1 = (const float*)d_in[3];
    const float* W2 = (const float*)d_in[4];  const float* b2 = (const float*)d_in[5];
    const float* W3 = (const float*)d_in[6];  const float* b3 = (const float*)d_in[7];
    const float* W4 = (const float*)d_in[8];  const float* b4 = (const float*)d_in[9];
    const float* W5 = (const float*)d_in[10]; const float* b5 = (const float*)d_in[11];

    const int n = in_sizes[0] / 256;   // 50000
    const int e = in_sizes[1] / 2;     // 800000
    const long long* src = ei;
    const long long* dst = ei + e;

    // workspace carving (256B aligned)
    char*  ws  = (char*)d_ws;
    size_t off = 0;
    auto carve = [&](size_t bytes) -> char* {
        char* p = ws + off;
        off += (bytes + 255) & ~(size_t)255;
        return p;
    };
    float* dinv  = (float*)carve((size_t)n * 4);
    float* sc    = (float*)carve((size_t)n * 4);
    float* wn    = (float*)carve((size_t)e * 4);
    unsigned short* Wt1 = (unsigned short*)carve(256 * 256 * 2);
    unsigned short* Wt2 = (unsigned short*)carve(256 * 256 * 2);
    unsigned short* Wt3 = (unsigned short*)carve(128 * 256 * 2);
    unsigned short* Wt4 = (unsigned short*)carve(64 * 128 * 2);
    unsigned short* Wt5 = (unsigned short*)carve(16 * 64 * 2);
    float* B0 = (float*)carve((size_t)n * 256 * 4);
    float* B1 = (float*)carve((size_t)n * 256 * 4);
    float* B2 = (float*)carve((size_t)n * 256 * 4);

    const int T = 256;
    auto blocks = [](long long total, int t) { return (unsigned)((total + t - 1) / t); };

    // ---- weight prep (bf16, transposed, padded) ----
    k_wconvert<<<blocks(256 * 256, T), T, 0, stream>>>(W1, Wt1, 256, 256, 256);
    k_wconvert<<<blocks(256 * 256, T), T, 0, stream>>>(W2, Wt2, 256, 256, 256);
    k_wconvert<<<blocks(128 * 256, T), T, 0, stream>>>(W3, Wt3, 256, 128, 128);
    k_wconvert<<<blocks(64 * 128, T),  T, 0, stream>>>(W4, Wt4, 128, 64, 64);
    k_wconvert<<<blocks(16 * 64, T),   T, 0, stream>>>(W5, Wt5, 64, 10, 16);

    // ---- graph norm (once, reused by all conv layers) ----
    k_deg_init <<<blocks(n, T), T, 0, stream>>>(dinv, n);
    k_deg_count<<<blocks(e, T), T, 0, stream>>>(dst, dinv, e);
    k_dinv     <<<blocks(n, T), T, 0, stream>>>(dinv, n);
    k_sc       <<<blocks(n, T), T, 0, stream>>>(dinv, sc, n);
    k_wnorm    <<<blocks(e, T), T, 0, stream>>>(dinv, src, dst, wn, e);

    const int rowTiles = (n + 15) / 16;
    const unsigned gx = (unsigned)((rowTiles + 7) / 8);

    // ---- layer 1: x(256) -> 256 ----
    gcn_gemm_wmma<4, 0><<<dim3(gx, 4), 256, 0, stream>>>(x, Wt1, B0, nullptr, n, 256, 256, 256);
    k_agg_init <<<blocks((long long)n * 64, T), T, 0, stream>>>(B0, sc, b1, B1, n, 256);
    k_agg_edges<<<blocks((long long)e * 64, T), T, 0, stream>>>(B0, wn, src, dst, B1, e, 256);

    // ---- layer 2: 256 -> 256 (lrelu fused into A load) ----
    gcn_gemm_wmma<4, 1><<<dim3(gx, 4), 256, 0, stream>>>(B1, Wt2, B2, nullptr, n, 256, 256, 256);
    k_agg_init <<<blocks((long long)n * 64, T), T, 0, stream>>>(B2, sc, b2, B0, n, 256);
    k_agg_edges<<<blocks((long long)e * 64, T), T, 0, stream>>>(B2, wn, src, dst, B0, e, 256);

    // ---- layer 3: 256 -> 128 ----
    gcn_gemm_wmma<4, 1><<<dim3(gx, 2), 256, 0, stream>>>(B0, Wt3, B1, nullptr, n, 256, 128, 128);
    k_agg_init <<<blocks((long long)n * 32, T), T, 0, stream>>>(B1, sc, b3, B2, n, 128);
    k_agg_edges<<<blocks((long long)e * 32, T), T, 0, stream>>>(B1, wn, src, dst, B2, e, 128);

    // ---- layer 4: 128 -> 64 ----
    gcn_gemm_wmma<4, 1><<<dim3(gx, 1), 256, 0, stream>>>(B2, Wt4, B0, nullptr, n, 128, 64, 64);
    k_agg_init <<<blocks((long long)n * 16, T), T, 0, stream>>>(B0, sc, b4, B1, n, 64);
    k_agg_edges<<<blocks((long long)e * 16, T), T, 0, stream>>>(B0, wn, src, dst, B1, e, 64);

    // ---- final: lrelu + dropout(0.5)*2 fused, 64 -> 10 (padded 16), bias fused ----
    gcn_gemm_wmma<1, 2><<<dim3(gx, 1), 256, 0, stream>>>(B1, Wt5, (float*)d_out, b5, n, 64, 10, 10);
}
